// FSDPAWindownAttnMask_10900626997585
// MI455X (gfx1250) — matrix operand order repaired and checked by reading, not compiled, not run on hardware
//
#include <hip/hip_runtime.h>
#include <stdint.h>
#include <stddef.h>

// ---------------------------------------------------------------------------
// MI455X (gfx1250) implementation.
//  K0: fp32 -> bf16 converts (x, qkv_w, proj_w)
//  K1: QKV GEMM   32768x3840x1280  bf16 WMMA, TDM tile staging, bf16 out (ws)
//  K2: per-(b,n,w) 16x16 head-axis attention, WMMA 16x16x32 bf16, bf16 out (ws)
//  K3: Proj GEMM  8192x1280x5120   bf16 WMMA, TDM tile staging, fp32 out+bias
// ---------------------------------------------------------------------------

typedef __attribute__((ext_vector_type(16))) __bf16        v16bf;
typedef __attribute__((ext_vector_type(8)))  float         v8f;
typedef __attribute__((ext_vector_type(4)))  unsigned int  v4u;
typedef __attribute__((ext_vector_type(8)))  int           v8i;
typedef __attribute__((ext_vector_type(4)))  int           v4i;
typedef __attribute__((ext_vector_type(4)))  unsigned int  u32x4;

// ---------------- TDM: 2D tile load global -> LDS (bf16 elements) ----------
// D# per CDNA5 ISA ch.8: group0 = {count/lds_addr/global_addr/type},
// group1 = {data_size, tensor dims, tile dims, stride}, groups 2/3 unused (2D).
__device__ __forceinline__ void tdm_load_tile_bf16(
    const void* gptr, uint32_t lds_off,
    uint32_t tile_k, uint32_t tile_rows,
    uint32_t tensor_k, uint32_t tensor_rows, uint32_t row_stride)
{
  uint64_t ga = (uint64_t)(uintptr_t)gptr;
  v4u g0;
  g0.x = 1u;                                   // count=1, user descriptor
  g0.y = lds_off;                              // lds_addr [63:32]
  g0.z = (uint32_t)ga;                         // global_addr low
  g0.w = (uint32_t)((ga >> 32) & 0x01FFFFFFu)  // global_addr[56:32]
       | (2u << 30);                           // type=2 ("image")
  v8i g1;
  g1[0] = (int)(1u << 16);                                        // data_size=1 (2B), wg_mask=0
  g1[1] = (int)((tensor_k & 0xFFFFu) << 16);                      // tensor_dim0[15:0]
  g1[2] = (int)(((tensor_k >> 16) & 0xFFFFu) |
                ((tensor_rows & 0xFFFFu) << 16));                 // dim0 hi | dim1 lo
  g1[3] = (int)(((tensor_rows >> 16) & 0xFFFFu) |
                ((tile_k & 0xFFFFu) << 16));                      // dim1 hi | tile_dim0
  g1[4] = (int)(tile_rows & 0xFFFFu);                             // tile_dim1 (tile_dim2=0)
  g1[5] = (int)row_stride;                                        // tensor_dim0_stride lo32
  g1[6] = 0;                                                      // stride hi / dim1_stride
  g1[7] = 0;
  v4i z4 = {0, 0, 0, 0};
#if __clang_major__ >= 23
  v8i z8 = {0, 0, 0, 0, 0, 0, 0, 0};
  __builtin_amdgcn_tensor_load_to_lds(g0, g1, z4, z4, z8, 0);
#else
  __builtin_amdgcn_tensor_load_to_lds(g0, g1, z4, z4, 0);
#endif
}

// ---------------- fp32 -> bf16 convert ------------------------------------
__global__ __launch_bounds__(256) void cvt_f32_bf16(
    const float* __restrict__ src, __bf16* __restrict__ dst, int n)
{
  int i = blockIdx.x * 256 + threadIdx.x;
  if (i < n) dst[i] = (__bf16)src[i];
}

// ---------------- bf16 WMMA GEMM with TDM staging -------------------------
// C[M,N] = A[M,K] * B[N,K]^T + bias ; A,B bf16 row-major over K.
// Block: 256 thr (8 waves), tile 128x128, K-step 32, double-buffered LDS.
template <int M, int N, int K, bool OUT_BF16>
__global__ __launch_bounds__(256) void gemm_bf16_wmma(
    const __bf16* __restrict__ A, const __bf16* __restrict__ B,
    const float* __restrict__ bias, void* __restrict__ Cout)
{
  constexpr int BM = 128, BN = 128, BK = 32;
  static_assert(M % BM == 0 && N % BN == 0 && K % BK == 0, "tile divisibility");
  __shared__ __bf16 As[2][BM][BK];   // 16 KB
  __shared__ __bf16 Bs[2][BN][BK];   // 16 KB

  const int nblk = N / BN;
  const int m0 = (blockIdx.x / nblk) * BM;
  const int n0 = (blockIdx.x % nblk) * BN;

  const int tid  = threadIdx.x;
  const int wid  = tid >> 5;
  const int lane = tid & 31;
  const int ln   = lane & 15;
  const int hi   = lane >> 4;

  const int wm = (wid >> 2) * 64;    // wave M offset (2 rows of waves)
  const int wn = (wid & 3) * 32;     // wave N offset (4 cols of waves)

  v8f acc[4][2];
#pragma unroll
  for (int i = 0; i < 4; ++i)
#pragma unroll
    for (int j = 0; j < 2; ++j) acc[i][j] = v8f{};

  if (wid == 0) {
    tdm_load_tile_bf16(A + (size_t)m0 * K, (uint32_t)(uintptr_t)&As[0][0][0],
                       BK, BM, K, M, K);
    tdm_load_tile_bf16(B + (size_t)n0 * K, (uint32_t)(uintptr_t)&Bs[0][0][0],
                       BK, BN, K, N, K);
  }

  const int KT = K / BK;
  for (int kt = 0; kt < KT; ++kt) {
    const int p = kt & 1;
    __builtin_amdgcn_s_wait_tensorcnt(0);   // buffer p fully resident
    __syncthreads();                        // everyone done with buffer p^1
    if (wid == 0 && kt + 1 < KT) {          // prefetch next tile (overlaps math)
      const int kn = (kt + 1) * BK;
      tdm_load_tile_bf16(A + (size_t)m0 * K + kn,
                         (uint32_t)(uintptr_t)&As[p ^ 1][0][0], BK, BM, K, M, K);
      tdm_load_tile_bf16(B + (size_t)n0 * K + kn,
                         (uint32_t)(uintptr_t)&Bs[p ^ 1][0][0], BK, BN, K, N, K);
    }

    v16bf afr[4], bfr[2];
#pragma unroll
    for (int tm = 0; tm < 4; ++tm) {        // A-frag: lane holds M-row, two 16B chunks
      const __bf16* pa = &As[p][wm + tm * 16 + ln][hi * 8];
      *(u32x4*)&afr[tm]       = *(const u32x4*)pa;        // k: hi*8 .. hi*8+7
      *(((u32x4*)&afr[tm])+1) = *(const u32x4*)(pa + 16); // k: hi*8+16 .. +23
    }
#pragma unroll
    for (int tn = 0; tn < 2; ++tn) {        // B-frag: lane holds N-column, two 16B chunks
      const __bf16* pb = &Bs[p][wn + tn * 16 + ln][hi * 8];
      *(u32x4*)&bfr[tn]       = *(const u32x4*)pb;
      *(((u32x4*)&bfr[tn])+1) = *(const u32x4*)(pb + 16);
    }
#pragma unroll
    for (int tm = 0; tm < 4; ++tm)
#pragma unroll
      for (int tn = 0; tn < 2; ++tn)
        acc[tm][tn] = __builtin_amdgcn_wmma_f32_16x16x32_bf16(
            false, afr[tm], false, bfr[tn], (short)0, acc[tm][tn], false, false);
  }

  // Store: lane holds col n = ln; VGPR v holds row v + 8*hi.
#pragma unroll
  for (int tn = 0; tn < 2; ++tn) {
    const int n = n0 + wn + tn * 16 + ln;
    const float bv = bias[n];
#pragma unroll
    for (int tm = 0; tm < 4; ++tm) {
#pragma unroll
      for (int v = 0; v < 8; ++v) {
        const int m = m0 + wm + tm * 16 + v + 8 * hi;
        const float val = acc[tm][tn][v] + bv;
        if constexpr (OUT_BF16)
          ((__bf16*)Cout)[(size_t)m * N + n] = (__bf16)val;
        else
          ((float*)Cout)[(size_t)m * N + n] = val;
      }
    }
  }
}

// ---------------- attention: one 16x16-over-heads unit per wave -----------
// qkv row u = s*8+b holds Q(16x80) | K(16x80) | V(16x80) contiguous (bf16).
__global__ __launch_bounds__(128) void attn_wmma(
    const __bf16* __restrict__ qkv, __bf16* __restrict__ outA)
{
  __shared__ __bf16 Pl[4][16][32];   // softmax probs, cols 16..31 zero-padded
  __shared__ __bf16 Vt[4][80][32];   // V^T, cols (g) 16..31 zero-padded

  const int tid  = threadIdx.x;
  const int ws   = tid >> 5;
  const int lane = tid & 31;
  const int ln   = lane & 15;
  const int hi   = lane >> 4;

  const int u = blockIdx.x * 4 + ws;       // 0..32767
  const int b = u & 7;
  const int s = u >> 3;
  const int w = s & 63;
  const int nw = s >> 6;

  const __bf16* Qb = qkv + (size_t)u * 3840;
  const __bf16* Kb = Qb + 1280;
  const __bf16* Vb = Qb + 2560;

  { // zero this wave's LDS slots
    u32x4 z = (u32x4)0u;
    u32x4* v0 = (u32x4*)&Vt[ws][0][0];
    for (int i = lane; i < 320; i += 32) v0[i] = z;   // 5120 B
    u32x4* p0 = (u32x4*)&Pl[ws][0][0];
    for (int i = lane; i < 64; i += 32) p0[i] = z;    // 1024 B
  }
  // V^T into LDS: Vt[d][g] = V[g][d]  (coalesced global reads)
#pragma unroll
  for (int g = 0; g < 16; ++g) {
#pragma unroll
    for (int j = 0; j < 3; ++j) {
      const int d = j * 32 + lane;
      if (d < 80) Vt[ws][d][g] = Vb[g * 80 + d];
    }
  }

  // S = Q * K^T over d=80 (padded to 96): 3 WMMAs
  v8f sacc = v8f{};
#pragma unroll
  for (int kt = 0; kt < 3; ++kt) {
    const int k0 = kt * 32;
    v16bf af, bf;
    { // Q A-frag: row h=ln, k chunks [k0+hi*8, +8) and [k0+hi*8+16, +8)
      const int d0 = k0 + hi * 8;
      *(u32x4*)&af = *(const u32x4*)(Qb + ln * 80 + d0);          // d0 <= 72
      const int d1 = d0 + 16;
      if (d1 < 80) *(((u32x4*)&af)+1) = *(const u32x4*)(Qb + ln * 80 + d1);
      else         *(((u32x4*)&af)+1) = (u32x4)0u;
    }
    { // K^T B-frag: col g=ln, k chunks [k0+hi*8, +8) and [k0+hi*8+16, +8)
      const int d0 = k0 + hi * 8;
      *(u32x4*)&bf = *(const u32x4*)(Kb + ln * 80 + d0);
      if (d0 + 16 < 80) {
        *(((u32x4*)&bf)+1) = *(const u32x4*)(Kb + ln * 80 + d0 + 16);
      } else {
        *(((u32x4*)&bf)+1) = (u32x4)0u;
      }
    }
    sacc = __builtin_amdgcn_wmma_f32_16x16x32_bf16(
        false, af, false, bf, (short)0, sacc, false, false);
  }

  // softmax over g (16 columns == 16 lanes of a half-wave), rows v+8*hi
  const float scale = 0.11180339887498949f;   // 1/sqrt(80)
  float p[8];
#pragma unroll
  for (int v = 0; v < 8; ++v) {
    float r = sacc[v] * scale;
    float mx = r;
#pragma unroll
    for (int off = 1; off < 16; off <<= 1)
      mx = fmaxf(mx, __shfl_xor(mx, off, 32));
    float e = __expf(r - mx);
    float sm = e;
#pragma unroll
    for (int off = 1; off < 16; off <<= 1)
      sm += __shfl_xor(sm, off, 32);
    p[v] = e / sm;
  }
#pragma unroll
  for (int v = 0; v < 8; ++v)
    Pl[ws][v + 8 * hi][ln] = (__bf16)p[v];
  __syncthreads();

  // O = P(16x16, padded K=32) * V(16x80): 5 WMMAs over d-tiles of 16
  v16bf pf;
  {
    const __bf16* pp = &Pl[ws][ln][hi * 8];
    *(u32x4*)&pf       = *(const u32x4*)pp;          // g: hi*8..+7
    *(((u32x4*)&pf)+1) = *(const u32x4*)(pp + 16);   // g: 16+.. (zeros)
  }
#pragma unroll
  for (int nt = 0; nt < 5; ++nt) {
    v16bf vf;
    const __bf16* pv = &Vt[ws][nt * 16 + ln][hi * 8];
    *(u32x4*)&vf       = *(const u32x4*)pv;          // g: hi*8..+7
    *(((u32x4*)&vf)+1) = *(const u32x4*)(pv + 16);   // g: 16+.. (zeros)
    v8f o = __builtin_amdgcn_wmma_f32_16x16x32_bf16(
        false, pf, false, vf, (short)0, v8f{}, false, false);
    const int d = nt * 16 + ln;
#pragma unroll
    for (int v = 0; v < 8; ++v) {
      const int h = v + 8 * hi;
      const size_t row = (size_t)((nw * 16 + h) * 8 + b);
      outA[row * 5120 + w * 80 + d] = (__bf16)o[v];
    }
  }
}

// ---------------------------------------------------------------------------
extern "C" void kernel_launch(void* const* d_in, const int* in_sizes, int n_in,
                              void* d_out, int out_size, void* d_ws, size_t ws_size,
                              hipStream_t stream) {
  (void)in_sizes; (void)n_in; (void)out_size; (void)ws_size;
  const float* x      = (const float*)d_in[0];
  // d_in[1] = cu_seqlens (ignored by the module)
  const float* qkv_w  = (const float*)d_in[2];
  const float* qkv_b  = (const float*)d_in[3];
  const float* proj_w = (const float*)d_in[4];
  const float* proj_b = (const float*)d_in[5];
  float* out = (float*)d_out;

  // workspace layout (bytes, all 256-aligned)
  char* ws = (char*)d_ws;
  __bf16* xbf    = (__bf16*)(ws);                         // 32768*1280
  __bf16* wqkvbf = (__bf16*)(ws + 83886080);              // 3840*1280
  __bf16* wprjbf = (__bf16*)(ws + 93716480);              // 1280*5120
  __bf16* qkvws  = (__bf16*)(ws + 106823680);             // 32768*3840
  __bf16* attnws = (__bf16*)(ws + 358481920);             // 8192*5120

  // K0: converts
  cvt_f32_bf16<<<(41943040 + 255) / 256, 256, 0, stream>>>(x, xbf, 41943040);
  cvt_f32_bf16<<<(4915200  + 255) / 256, 256, 0, stream>>>(qkv_w,  wqkvbf, 4915200);
  cvt_f32_bf16<<<(6553600  + 255) / 256, 256, 0, stream>>>(proj_w, wprjbf, 6553600);

  // K1: qkv = x @ qkv_w^T + qkv_b   (bf16 out to ws)
  gemm_bf16_wmma<32768, 3840, 1280, true>
      <<<(32768 / 128) * (3840 / 128), 256, 0, stream>>>(xbf, wqkvbf, qkv_b, qkvws);

  // K2: windowed head-axis attention (one 16x16 unit per wave)
  attn_wmma<<<32768 / 4, 128, 0, stream>>>(qkvws, attnws);

  // K3: out = attn @ proj_w^T + proj_b   (fp32 out)
  gemm_bf16_wmma<8192, 1280, 5120, false>
      <<<(8192 / 128) * (1280 / 128), 256, 0, stream>>>(attnws, wprjbf, proj_b, out);
}
